// QuadraticHeterogenousCrossAttention_82094004895925
// MI455X (gfx1250) — matrix operand order repaired and checked by reading, not compiled, not run on hardware
//
#include <hip/hip_runtime.h>
#include <hip/hip_bf16.h>

typedef _Float16 half_t;
typedef __attribute__((ext_vector_type(16))) _Float16 v16h;
typedef __attribute__((ext_vector_type(2)))  _Float16 v2h;
typedef __attribute__((ext_vector_type(8)))  float    v8f;

#define DD    512
#define SS    1024
#define HH    8
#define DHH   64
#define BB    4
#define FF    4
#define MTOT  4096            // BB*SS
#define WSZ   (512*512)
#define BHSD  ((size_t)BB*HH*SS*DHH)
#define NCH   (SS / 32)       // key chunks per sequence

// ---------------------------------------------------------------- helpers

__device__ inline v8f wmma16(v16h a, v16h b, v8f c) {
  // D = A(16x32 f16) * B(32x16 f16) + C(16x16 f32)
  return __builtin_amdgcn_wmma_f32_16x16x32_f16(false, a, false, b, (short)0, c,
                                                false, false);
}

__device__ inline v8f vzero8() {
  v8f z;
#pragma unroll
  for (int i = 0; i < 8; ++i) z[i] = 0.0f;
  return z;
}

// A-matrix (16 rows x 32 K) from LDS, storage row-major base[row*ldr + k].
// Per-lane halves are two contiguous 16B runs -> 2x ds_load_b128.
__device__ inline v16h lds_load_A(const half_t* base, int ldr, int lane) {
  int row  = lane & 15;
  int koff = (lane >> 4) << 4;       // 0 or 16
  v16h a;
#pragma unroll
  for (int j = 0; j < 16; ++j) {
    int k = j + koff;
    a[j] = base[row * ldr + k];
  }
  return a;
}

// B-matrix (32 K x 16 N), storage base[n*ldr + k] (N-major).
// Per-lane halves are one contiguous 32B run -> 2x ds_load_b128.
__device__ inline v16h lds_load_Bt(const half_t* base, int ldr, int lane) {
  int n    = lane & 15;
  int koff = (lane >> 4) << 4;       // 0 or 16
  v16h b;
#pragma unroll
  for (int j = 0; j < 16; ++j) b[j] = base[n * ldr + koff + j];
  return b;
}

__device__ inline float redmax16(float v) {
#pragma unroll
  for (int off = 1; off < 16; off <<= 1) v = fmaxf(v, __shfl_xor(v, off, 32));
  return v;
}
__device__ inline float redsum16(float v) {
#pragma unroll
  for (int off = 1; off < 16; off <<= 1) v += __shfl_xor(v, off, 32);
  return v;
}

// CDNA5 async global->LDS copy (16B per lane), tracked by ASYNCcnt.
__device__ inline void async_load_b128(unsigned lds_off, unsigned gl_off,
                                       const void* saddr) {
  asm volatile("global_load_async_to_lds_b128 %0, %1, %2"
               :: "v"(lds_off), "v"(gl_off), "s"(saddr)
               : "memory");
}
__device__ inline void wait_async0() {
  asm volatile("s_wait_asynccnt 0x0" ::: "memory");
}
__device__ inline void wait_async2() {
  // async loads complete in order: ASYNCcnt<=2 means the older chunk's
  // two loads have landed while the just-issued pair is still in flight.
  asm volatile("s_wait_asynccnt 0x2" ::: "memory");
}

__device__ inline unsigned lds_off_of(const void* p) {
  return (unsigned)(uintptr_t)p;      // low 32 bits of generic LDS addr
}

// ---------------------------------------------------------------- kernel 1
// Convert all projection weights to TRANSPOSED f16 (W^T[n][k]); softmax gates.
__global__ __launch_bounds__(256) void cvt_weights_kernel(
    const float* __restrict__ Wq, const float* __restrict__ Wk,
    const float* __restrict__ Wv, const float* __restrict__ Wo,
    const float* __restrict__ fw,
    half_t* __restrict__ WqT, half_t* __restrict__ WkT,
    half_t* __restrict__ WvT, half_t* __restrict__ WoT,
    float* __restrict__ gates) {
  size_t idx = (size_t)blockIdx.x * 256 + threadIdx.x;
  size_t r = idx / WSZ, o = idx % WSZ;
  size_t k = o >> 9, n = o & 511;          // source W[k][n]
  size_t to = n * DD + k;                  // dest  W^T[n][k]
  if (r == 0)      WqT[to] = (half_t)Wq[o];
  else if (r <= 4) { size_t f = r - 1; WkT[f*WSZ + to] = (half_t)Wk[f*WSZ + o]; }
  else if (r <= 8) { size_t f = r - 5; WvT[f*WSZ + to] = (half_t)Wv[f*WSZ + o]; }
  else if (r == 9) WoT[to] = (half_t)Wo[o];
  if (idx == 0) {
    float m = fmaxf(fmaxf(fw[0], fw[1]), fmaxf(fw[2], fw[3]));
    float e0 = __expf(fw[0]-m), e1 = __expf(fw[1]-m);
    float e2 = __expf(fw[2]-m), e3 = __expf(fw[3]-m);
    float s = e0 + e1 + e2 + e3;
    gates[0] = e0/s; gates[1] = e1/s; gates[2] = e2/s; gates[3] = e3/s;
  }
}

// ---------------------------------------------------------------- kernel 2
// Q/K/V projection GEMMs (f32 src @ f16 W^T, f32 acc), fused bias + RoPE.
// blockIdx.z: 0 = Q, 1..4 = K[f], 5..8 = V[f].
// Q,K stored [b][h][s][e]; V stored [b][h][e][s] (feeds attention directly).
__global__ __launch_bounds__(256) void proj_kernel(
    const float* __restrict__ q_data, const float* __restrict__ x_0,
    const float* __restrict__ v_0,    const float* __restrict__ edge,
    const float* __restrict__ cfeat,
    const half_t* __restrict__ WqT, const half_t* __restrict__ WkT,
    const half_t* __restrict__ WvT,
    const float* __restrict__ bq, const float* __restrict__ bk,
    const float* __restrict__ bv,
    half_t* __restrict__ Q16, half_t* __restrict__ K16,
    half_t* __restrict__ V16) {
  __shared__ __align__(16) half_t As[32 * 40];    // 32 rows x 32 K
  __shared__ __align__(16) half_t BsT[64 * 40];   // 64 N x 32 K (N-major)

  int tid = threadIdx.x, wave = tid >> 5, lane = tid & 31;
  int lane16 = lane & 15, hi = lane >> 4;
  int wm = wave >> 2, wn = wave & 3;              // 2 x 4 wave tiling
  int m0 = blockIdx.x * 32, n0 = blockIdx.y * 64;

  const float* feats[4] = {x_0, v_0, edge, cfeat};
  int p = blockIdx.z;
  const float* src; const half_t* w; const float* bias; half_t* dst;
  int dorope, vmode = 0; float scale = 1.0f;
  if (p == 0) {
    src = q_data; w = WqT; bias = bq; dst = Q16; dorope = 1; scale = 0.125f;
  } else if (p <= 4) {
    int f = p - 1;
    src = feats[f]; w = WkT + (size_t)f * WSZ; bias = bk + f * DD;
    dst = K16 + (size_t)f * BHSD; dorope = 1;
  } else {
    int f = p - 5;
    src = feats[f]; w = WvT + (size_t)f * WSZ; bias = bv + f * DD;
    dst = V16 + (size_t)f * BHSD; dorope = 0; vmode = 1;
  }

  v8f acc = vzero8();
  for (int k0 = 0; k0 < DD; k0 += 32) {
    __syncthreads();
    // stage A tile: f32 -> f16, packed half2 stores
#pragma unroll
    for (int i = 0; i < 2; ++i) {
      int seg = tid + i * 256;                    // 512 half2 segments
      int row = seg >> 4, cp = seg & 15;
      float2 fv = *(const float2*)&src[(size_t)(m0 + row) * DD + k0 + cp * 2];
      v2h hv; hv[0] = (half_t)fv.x; hv[1] = (half_t)fv.y;
      *(v2h*)&As[row * 40 + cp * 2] = hv;
    }
    // stage B tile from W^T: coalesced uint4 copy (N-major)
    {
      int row = tid >> 2, cs = tid & 3;           // 64 rows x 4 segs
      *(uint4*)&BsT[row * 40 + cs * 8] =
          *(const uint4*)&w[(size_t)(n0 + row) * DD + k0 + cs * 8];
    }
    __syncthreads();
    v16h a = lds_load_A(&As[(wm * 16) * 40], 40, lane);
    v16h b = lds_load_Bt(&BsT[(wn * 16) * 40], 40, lane);
    acc = wmma16(a, b, acc);
  }

  // epilogue: bias + time-only RoPE + f16 scatter
  int c = n0 + wn * 16 + lane16;                  // global output column
  float bval = bias[c];
  int h = c >> 6, e = c & 63;
  float freq = 0.0f;
  if (dorope) {
    int i = e >> 1;                               // same for both pair lanes
    freq = __expf(-(float)i * (6.907755278982137f / 32.0f)); // 1000^(-i/32)
  }
#pragma unroll
  for (int r = 0; r < 8; ++r) {
    int m = m0 + wm * 16 + r + hi * 8;            // global row (b*1024 + s)
    float v = acc[r] + bval;
    int s = m & (SS - 1);
    if (dorope) {
      float partner = __shfl_xor(v, 1, 32);
      float t = (float)(s >> 7);                  // time index = s // N
      float ang = t * freq;
      float cs = __cosf(ang), sn = __sinf(ang);
      v = (e & 1) ? (partner * sn + v * cs) : (v * cs - partner * sn);
    }
    v *= scale;
    int b_ = m >> 10;
    size_t bh = (size_t)(b_ * HH + h);
    size_t idx = vmode ? (bh * DHH + e) * SS + s : (bh * SS + s) * DHH + e;
    dst[idx] = (half_t)v;
  }
}

// ---------------------------------------------------------------- kernel 3
// Flash attention over 4 features with gated accumulation.
// grid: (S/128, H, B); block: 256 threads = 8 waves, 16 query rows per wave.
// K/V tiles double-buffered through the CDNA5 async-to-LDS path.
__global__ __launch_bounds__(256) void attn_kernel(
    const half_t* __restrict__ Q16, const half_t* __restrict__ K16,
    const half_t* __restrict__ V16, const float* __restrict__ gates,
    half_t* __restrict__ AttnOut) {
  __shared__ __align__(16) half_t Qs[128 * 72];      // [q][e]
  __shared__ __align__(16) half_t Ks[2][32 * 72];    // [key][e], 2 buffers
  __shared__ __align__(16) half_t Vts[2][64 * 40];   // [e][key], 2 buffers
  __shared__ __align__(16) half_t Ps[8 * 16 * 40];   // per-wave P scratch

  int tid = threadIdx.x, wave = tid >> 5, lane = tid & 31;
  int lane16 = lane & 15, hi = lane >> 4;
  int qblk = blockIdx.x, h = blockIdx.y, b = blockIdx.z;
  size_t bh = (size_t)b * HH + h;
  int q0 = qblk * 128;

  const half_t* Qg = Q16 + bh * SS * DHH;
#pragma unroll
  for (int i = 0; i < 4; ++i) {
    int seg = tid + i * 256;                      // 1024 uint4 segments
    int row = seg >> 3, cs = seg & 7;
    *(uint4*)&Qs[row * 72 + cs * 8] =
        *(const uint4*)&Qg[(size_t)(q0 + row) * DHH + cs * 8];
  }
  __syncthreads();

  v16h aq0 = lds_load_A(&Qs[(wave * 16) * 72], 72, lane);        // e 0..31
  v16h aq1 = lds_load_A(&Qs[(wave * 16) * 72 + 32], 72, lane);   // e 32..63
  half_t* Pw = &Ps[wave * 16 * 40];

  // async-staging addresses (fixed per thread, one per buffer)
  int krow = tid >> 3, kcs = tid & 7;             // K: 32 rows x 8 segs
  int vrow = tid >> 2, vcs = tid & 3;             // V: 64 rows x 4 segs
  unsigned ldsK[2], ldsV[2];
#pragma unroll
  for (int bi = 0; bi < 2; ++bi) {
    ldsK[bi] = lds_off_of(&Ks[bi][krow * 72 + kcs * 8]);
    ldsV[bi] = lds_off_of(&Vts[bi][vrow * 40 + vcs * 8]);
  }
  unsigned kgoff = (unsigned)(krow * DHH + kcs * 8) * 2;
  unsigned vgoff = (unsigned)(vrow * SS + vcs * 8) * 2;

  v8f accF[4];
#pragma unroll
  for (int t = 0; t < 4; ++t) accF[t] = vzero8();

  for (int f = 0; f < FF; ++f) {
    const half_t* Kg = K16 + ((size_t)f * BB * HH + bh) * SS * DHH;
    const half_t* Vg = V16 + ((size_t)f * BB * HH + bh) * DHH * SS; // [e][s]
    v8f acco[4];
#pragma unroll
    for (int t = 0; t < 4; ++t) acco[t] = vzero8();
    float mrows[8], lrows[8];
#pragma unroll
    for (int r = 0; r < 8; ++r) { mrows[r] = -3.0e38f; lrows[r] = 0.0f; }

    // prologue: issue chunk 0 into buffer 0
    async_load_b128(ldsK[0], kgoff, Kg);
    async_load_b128(ldsV[0], vgoff, Vg);

    for (int kc = 0; kc < NCH; ++kc) {
      int cur = kc & 1;
      if (kc + 1 < NCH) {
        // buffer (1-cur) was released by the end-of-iteration barrier of
        // iteration kc-1, so the async engine may overwrite it now.
        async_load_b128(ldsK[1 - cur], (unsigned)((kc + 1) * 32 * DHH * 2) + kgoff, Kg);
        async_load_b128(ldsV[1 - cur], (unsigned)((kc + 1) * 32 * 2) + vgoff, Vg);
        wait_async2();                 // chunk kc landed; kc+1 in flight
      } else {
        wait_async0();
      }
      __syncthreads();                 // all waves' tiles visible
      const half_t* Kc = Ks[cur];
      const half_t* Vc = Vts[cur];

      // scores: Q(16x64) . K^T(64x32) -> two 16x16 f32 tiles
      v8f s0 = vzero8(), s1 = vzero8();
      s0 = wmma16(aq0, lds_load_Bt(&Kc[0], 72, lane), s0);
      s0 = wmma16(aq1, lds_load_Bt(&Kc[32], 72, lane), s0);
      s1 = wmma16(aq0, lds_load_Bt(&Kc[16 * 72], 72, lane), s1);
      s1 = wmma16(aq1, lds_load_Bt(&Kc[16 * 72 + 32], 72, lane), s1);

      // online softmax (row stats replicated across each 16-lane half)
      float corr[8];
#pragma unroll
      for (int r = 0; r < 8; ++r) {
        float rm = redmax16(fmaxf(s0[r], s1[r]));
        float mn = fmaxf(mrows[r], rm);
        corr[r] = __expf(mrows[r] - mn);
        float p0 = __expf(s0[r] - mn);
        float p1 = __expf(s1[r] - mn);
        s0[r] = p0; s1[r] = p1;
        lrows[r] = lrows[r] * corr[r] + redsum16(p0 + p1);
        mrows[r] = mn;
      }
#pragma unroll
      for (int t = 0; t < 4; ++t)
#pragma unroll
        for (int r = 0; r < 8; ++r) acco[t][r] *= corr[r];

      // re-layout P (16x32) into WMMA A format via per-wave LDS scratch
#pragma unroll
      for (int r = 0; r < 8; ++r) {
        int M = r + hi * 8;
        Pw[M * 40 + lane16]      = (half_t)s0[r];
        Pw[M * 40 + 16 + lane16] = (half_t)s1[r];
      }
      v16h ap = lds_load_A(Pw, 40, lane);
#pragma unroll
      for (int t = 0; t < 4; ++t)
        acco[t] = wmma16(ap, lds_load_Bt(&Vc[(t * 16) * 40], 40, lane),
                         acco[t]);

      __syncthreads();                 // release buffer `cur` for overwrite
    }

    float g = gates[f];
#pragma unroll
    for (int r = 0; r < 8; ++r) {
      float inv = g / lrows[r];
#pragma unroll
      for (int t = 0; t < 4; ++t) accF[t][r] += acco[t][r] * inv;
    }
  }

  // store to [b*S + s][h*64 + e] f16 (the layout the out-proj consumes)
  size_t mbase = (size_t)b * SS + q0 + wave * 16;
#pragma unroll
  for (int t = 0; t < 4; ++t)
#pragma unroll
    for (int r = 0; r < 8; ++r) {
      size_t m = mbase + r + hi * 8;
      int c = h * DHH + t * 16 + lane16;
      AttnOut[m * DD + c] = (half_t)accF[t][r];
    }
}

// ---------------------------------------------------------------- kernel 4
// Output projection: AttnOut(f16) @ Wo^T + bo -> f32 d_out.
__global__ __launch_bounds__(256) void outproj_kernel(
    const half_t* __restrict__ A, const half_t* __restrict__ WoT,
    const float* __restrict__ bo, float* __restrict__ out) {
  __shared__ __align__(16) half_t As[32 * 40];
  __shared__ __align__(16) half_t BsT[64 * 40];

  int tid = threadIdx.x, wave = tid >> 5, lane = tid & 31;
  int lane16 = lane & 15, hi = lane >> 4;
  int wm = wave >> 2, wn = wave & 3;
  int m0 = blockIdx.x * 32, n0 = blockIdx.y * 64;

  v8f acc = vzero8();
  for (int k0 = 0; k0 < DD; k0 += 32) {
    __syncthreads();
    if (tid < 128) {                              // 32 rows x 4 uint4 segs
      int row = tid >> 2, cs = tid & 3;
      *(uint4*)&As[row * 40 + cs * 8] =
          *(const uint4*)&A[(size_t)(m0 + row) * DD + k0 + cs * 8];
    }
    {
      int row = tid >> 2, cs = tid & 3;           // 64 rows x 4 segs
      *(uint4*)&BsT[row * 40 + cs * 8] =
          *(const uint4*)&WoT[(size_t)(n0 + row) * DD + k0 + cs * 8];
    }
    __syncthreads();
    v16h a = lds_load_A(&As[(wm * 16) * 40], 40, lane);
    v16h b = lds_load_Bt(&BsT[(wn * 16) * 40], 40, lane);
    acc = wmma16(a, b, acc);
  }

  int c = n0 + wn * 16 + lane16;
  float bval = bo[c];
#pragma unroll
  for (int r = 0; r < 8; ++r) {
    int m = m0 + wm * 16 + r + hi * 8;
    out[(size_t)m * DD + c] = acc[r] + bval;
  }
}

// ---------------------------------------------------------------- launch

extern "C" void kernel_launch(void* const* d_in, const int* in_sizes, int n_in,
                              void* d_out, int out_size, void* d_ws,
                              size_t ws_size, hipStream_t stream) {
  (void)in_sizes; (void)n_in; (void)out_size; (void)ws_size;
  const float* q_data = (const float*)d_in[0];
  const float* x_0    = (const float*)d_in[1];
  const float* v_0    = (const float*)d_in[2];
  const float* edge   = (const float*)d_in[3];
  const float* cfeat  = (const float*)d_in[4];
  const float* Wq = (const float*)d_in[5];
  const float* bq = (const float*)d_in[6];
  const float* Wk = (const float*)d_in[7];
  const float* bk = (const float*)d_in[8];
  const float* Wv = (const float*)d_in[9];
  const float* bv = (const float*)d_in[10];
  const float* Wo = (const float*)d_in[11];
  const float* bo = (const float*)d_in[12];
  const float* fw = (const float*)d_in[13];
  float* out = (float*)d_out;

  char* ws = (char*)d_ws;
  size_t off = 0;
  auto carve = [&](size_t bytes) -> char* {
    char* p = ws + off;
    off += (bytes + 255) & ~(size_t)255;
    return p;
  };
  half_t* WqT = (half_t*)carve((size_t)WSZ * 2);
  half_t* WkT = (half_t*)carve((size_t)4 * WSZ * 2);
  half_t* WvT = (half_t*)carve((size_t)4 * WSZ * 2);
  half_t* WoT = (half_t*)carve((size_t)WSZ * 2);
  half_t* Q16 = (half_t*)carve(BHSD * 2);
  half_t* K16 = (half_t*)carve((size_t)FF * BHSD * 2);
  half_t* V16 = (half_t*)carve((size_t)FF * BHSD * 2);
  half_t* AO16 = (half_t*)carve((size_t)MTOT * DD * 2);
  float*  gates = (float*)carve(4 * sizeof(float));

  cvt_weights_kernel<<<(10 * WSZ) / 256, 256, 0, stream>>>(
      Wq, Wk, Wv, Wo, fw, WqT, WkT, WvT, WoT, gates);
  proj_kernel<<<dim3(MTOT / 32, DD / 64, 9), 256, 0, stream>>>(
      q_data, x_0, v_0, edge, cfeat, WqT, WkT, WvT, bq, bk, bv, Q16, K16, V16);
  attn_kernel<<<dim3(SS / 128, HH, BB), 256, 0, stream>>>(Q16, K16, V16, gates,
                                                          AO16);
  outproj_kernel<<<dim3(MTOT / 32, DD / 64), 256, 0, stream>>>(AO16, WoT, bo,
                                                               out);
}
